// GraphNetBlock_68753836474499
// MI455X (gfx1250) — matrix operand order repaired and checked by reading, not compiled, not run on hardware
//
#include <hip/hip_runtime.h>

#define H 256
#define NNODES 20000
#define NEDGES 320000
#define EPS 1e-5f

typedef __attribute__((ext_vector_type(16))) __bf16 v16bf;
typedef __attribute__((ext_vector_type(8)))  float  v8f;
typedef __attribute__((ext_vector_type(8)))  unsigned int v8u;
typedef unsigned short u16;
typedef unsigned int   u32;

__device__ __forceinline__ u16 f2bf(float x) {
  u32 u = __builtin_bit_cast(u32, x);
  u32 r = u + 0x7FFFu + ((u >> 16) & 1u);   // round-to-nearest-even
  return (u16)(r >> 16);
}

// SiLU via v_rcp_f32 (bf16-rounded pipeline -> hw rcp accuracy is plenty)
__device__ __forceinline__ float silu_f(float x) {
  return x * __builtin_amdgcn_rcpf(1.0f + __expf(-x));
}

// ---------------- weight conversion: f32 [K][N] -> bf16 transposed [N][K] ----
__global__ __launch_bounds__(256) void convert_weights_kernel(
    const float* __restrict__ Wsrc, const float* __restrict__ Wdst,
    const float* __restrict__ We,   const float* __restrict__ Wout,
    const float* __restrict__ Wn1,  const float* __restrict__ Wn2,
    u16* __restrict__ wcat_t, u16* __restrict__ wout_t,
    u16* __restrict__ wn1_t,  u16* __restrict__ wn2_t)
{
  int tid = blockIdx.x * 256 + threadIdx.x;
  if (tid < 196608) {                       // Wcat_t [256][768] = [Wsrc;Wdst;We]^T
    int n = tid / 768, k = tid % 768;
    float v = (k < 256) ? Wsrc[k * H + n]
            : (k < 512) ? Wdst[(k - 256) * H + n]
                        : We[(k - 512) * H + n];
    wcat_t[n * 768 + k] = f2bf(v);
  } else if (tid < 262144) {                // Wout_t [256][256]
    int t = tid - 196608; int n = t / 256, k = t % 256;
    wout_t[n * 256 + k] = f2bf(Wout[k * H + n]);
  } else if (tid < 393216) {                // Wn1_t [256][512]
    int t = tid - 262144; int n = t / 512, k = t % 512;
    wn1_t[n * 512 + k] = f2bf(Wn1[k * H + n]);
  } else if (tid < 458752) {                // Wn2_t [256][256]
    int t = tid - 393216; int n = t / 256, k = t % 256;
    wn2_t[n * 256 + k] = f2bf(Wn2[k * H + n]);
  }
}

__global__ __launch_bounds__(256) void convert_h_kernel(
    const float* __restrict__ h, u16* __restrict__ h_bf)
{
  int i = blockIdx.x * 256 + threadIdx.x;   // i < N*H/4
  float4 f = ((const float4*)h)[i];
  uint2 p;
  p.x = (u32)f2bf(f.x) | ((u32)f2bf(f.y) << 16);
  p.y = (u32)f2bf(f.z) | ((u32)f2bf(f.w) << 16);
  ((uint2*)h_bf)[i] = p;
}

// ---------------- shared GEMM tile engine -----------------------------------
// Block = 128 threads (4 waves). Output tile: 32 rows x 256 cols.
// Wave w owns cols [w*64, w*64+64) as 2(m-tiles) x 4(n-tiles) of 16x16 WMMA.
// A: LDS bf16 [32][ASTRIDE]; Bt: global bf16 transposed [256][KTOT];
// Bs: LDS staging [256][40] (row stride 80B -> b128-aligned, conflict-free).
#define BSTRIDE 40

template<int KTOT, int ASTRIDE>
__device__ __forceinline__ void gemm_block(
    const u16* A, const u16* __restrict__ Bt, u16* Bs,
    v8f acc[2][4], int tid)
{
  const int lane = tid & 31;
  const int wave = tid >> 5;
  const int m0 = lane & 15;
  const int hi = lane >> 4;
  for (int ks = 0; ks < KTOT; ks += 32) {
    // cooperative stage of the 32-K weight slab (256 rows x 64B)
#pragma unroll
    for (int it = 0; it < 8; ++it) {
      int linear = it * 128 + tid;
      int n = linear >> 2, part = linear & 3;
      *(uint4*)(Bs + n * BSTRIDE + part * 8) =
          *(const uint4*)(Bt + (size_t)n * KTOT + ks + part * 8);
    }
    __syncthreads();

    v16bf bfrag[4];
#pragma unroll
    for (int nt = 0; nt < 4; ++nt) {
      int n = wave * 64 + nt * 16 + m0;     // output column
      v8u u;
#pragma unroll
      for (int v = 0; v < 8; ++v)           // B 32x16: same K order as A (K=8*hi+0..7, 16+8*hi+0..7)
        u[v] = *(const u32*)(Bs + n * BSTRIDE + (v < 4 ? 2 * v : 16 + 2 * (v - 4)) + 8 * hi);
      bfrag[nt] = __builtin_bit_cast(v16bf, u);
    }
#pragma unroll
    for (int mt = 0; mt < 2; ++mt) {
      v8u u;
#pragma unroll
      for (int v = 0; v < 8; ++v) {         // A 16x32 bf16 lane layout (ISA 7.12.2)
        int kb = (v < 4 ? 2 * v : 16 + 2 * (v - 4)) + 8 * hi;
        u[v] = *(const u32*)(A + (mt * 16 + m0) * ASTRIDE + ks + kb);
      }
      v16bf afrag = __builtin_bit_cast(v16bf, u);
#pragma unroll
      for (int nt = 0; nt < 4; ++nt)
        acc[mt][nt] = __builtin_amdgcn_wmma_f32_16x16x32_bf16(
            false, afrag, false, bfrag[nt], (short)0, acc[mt][nt], false, false);
    }
    __syncthreads();
  }
}

// ---------------- edge kernel ------------------------------------------------
#define A1E_STRIDE 776
#define A1E_BYTES (32 * 776 * 2)   // 49664
#define BS_BYTES  (256 * 40 * 2)   // 20480
#define A2_STRIDE 264
#define A2_BYTES  (32 * 264 * 2)   // 16896

__global__ __launch_bounds__(128) void edge_kernel(
    const float* __restrict__ e, const int* __restrict__ eidx,
    const u16* __restrict__ h_bf,
    const u16* __restrict__ Wcat_t, const u16* __restrict__ Wout_t,
    const float* __restrict__ b_src, const float* __restrict__ b_out,
    const float* __restrict__ g_e, const float* __restrict__ beta_e,
    float* __restrict__ e_new, float* __restrict__ agg)
{
  __shared__ __align__(16) unsigned char smem[A1E_BYTES + BS_BYTES + A2_BYTES];
  u16* A1 = (u16*)smem;
  u16* Bs = (u16*)(smem + A1E_BYTES);
  u16* A2 = (u16*)(smem + A1E_BYTES + BS_BYTES);
  float* XB = (float*)smem;                 // aliases A1 (dead after GEMM1)
  __shared__ int sSrc[32], sDst[32];

  const int tid = threadIdx.x;
  const int e0 = blockIdx.x * 32;
  if (tid < 32) { sSrc[tid] = eidx[e0 + tid]; sDst[tid] = eidx[NEDGES + e0 + tid]; }
  __syncthreads();

  // A rows: [h_src(0..255) | h_dst(256..511) | e(512..767)] in bf16
#pragma unroll
  for (int it = 0; it < 16; ++it) {
    int linear = it * 128 + tid;            // 2048 chunks of 8 u16
    int r = linear >> 6, col = (linear & 63) * 8;
    const u16* sp = (col < 256) ? (h_bf + (size_t)sSrc[r] * H + col)
                                : (h_bf + (size_t)sDst[r] * H + (col - 256));
    *(uint4*)(A1 + r * A1E_STRIDE + col) = *(const uint4*)sp;
  }
#pragma unroll
  for (int it = 0; it < 16; ++it) {
    int linear = it * 128 + tid;            // 2048 chunks of 4 f32 -> bf16
    int r = linear >> 6, col = (linear & 63) * 4;
    float4 f = *(const float4*)(e + (size_t)(e0 + r) * H + col);
    u16* dp = A1 + r * A1E_STRIDE + 512 + col;
    dp[0] = f2bf(f.x); dp[1] = f2bf(f.y); dp[2] = f2bf(f.z); dp[3] = f2bf(f.w);
  }
  __syncthreads();

  const int lane = tid & 31, wave = tid >> 5;
  const int m0 = lane & 15, hi = lane >> 4;
  const v8f vzero = {0.f, 0.f, 0.f, 0.f, 0.f, 0.f, 0.f, 0.f};
  v8f acc[2][4];
#pragma unroll
  for (int mt = 0; mt < 2; ++mt)
#pragma unroll
    for (int nt = 0; nt < 4; ++nt) acc[mt][nt] = vzero;

  gemm_block<768, A1E_STRIDE>(A1, Wcat_t, Bs, acc, tid);

  // bias + SiLU -> A2 (bf16)
#pragma unroll
  for (int nt = 0; nt < 4; ++nt) {
    int n = wave * 64 + nt * 16 + m0;
    float bn = b_src[n];
#pragma unroll
    for (int mt = 0; mt < 2; ++mt)
#pragma unroll
      for (int j = 0; j < 8; ++j) {
        int m = mt * 16 + j + 8 * hi;       // C/D layout: VGPR j -> M=j (+8 for hi lanes)
        A2[m * A2_STRIDE + n] = f2bf(silu_f(acc[mt][nt][j] + bn));
      }
  }
  __syncthreads();

#pragma unroll
  for (int mt = 0; mt < 2; ++mt)
#pragma unroll
    for (int nt = 0; nt < 4; ++nt) acc[mt][nt] = vzero;
  gemm_block<256, A2_STRIDE>(A2, Wout_t, Bs, acc, tid);

  // de + b_out + residual(e) -> XB (f32)
#pragma unroll
  for (int nt = 0; nt < 4; ++nt) {
    int n = wave * 64 + nt * 16 + m0;
    float bn = b_out[n];
#pragma unroll
    for (int mt = 0; mt < 2; ++mt)
#pragma unroll
      for (int j = 0; j < 8; ++j) {
        int m = mt * 16 + j + 8 * hi;
        XB[m * H + n] = acc[mt][nt][j] + bn + e[(size_t)(e0 + m) * H + n];
      }
  }
  __syncthreads();

  // LayerNorm (4 threads per row, shuffle reduce), write e_new, scatter into agg
  const int r = tid >> 2, cb = (tid & 3) * 64;
  float s = 0.f, ss = 0.f;
  const float* xr = XB + r * H;
#pragma unroll 8
  for (int c = 0; c < 64; ++c) { float x = xr[cb + c]; s += x; ss += x * x; }
  s  += __shfl_xor(s, 1, 32);  s  += __shfl_xor(s, 2, 32);
  ss += __shfl_xor(ss, 1, 32); ss += __shfl_xor(ss, 2, 32);
  float mu = s * (1.f / H);
  float rstd = __builtin_amdgcn_rsqf(ss * (1.f / H) - mu * mu + EPS);
  float* aggrow = agg + (size_t)sDst[r] * H;
  float* orow = e_new + (size_t)(e0 + r) * H;
#pragma unroll 4
  for (int c = 0; c < 64; ++c) {
    int cc = cb + c;
    float y = (xr[cc] - mu) * rstd * g_e[cc] + beta_e[cc];
    orow[cc] = y;
    __hip_atomic_fetch_add(aggrow + cc, y, __ATOMIC_RELAXED, __HIP_MEMORY_SCOPE_AGENT);
  }
}

// ---------------- node kernel ------------------------------------------------
#define A1N_STRIDE 520
#define A1N_BYTES (32 * 520 * 2)   // 33280

__global__ __launch_bounds__(128) void node_kernel(
    const float* __restrict__ h, const u16* __restrict__ h_bf,
    const float* __restrict__ agg,
    const u16* __restrict__ Wn1_t, const u16* __restrict__ Wn2_t,
    const float* __restrict__ b_n1, const float* __restrict__ b_n2,
    const float* __restrict__ g_n, const float* __restrict__ beta_n,
    float* __restrict__ h_new)
{
  __shared__ __align__(16) unsigned char smem[A1N_BYTES + BS_BYTES + A2_BYTES];
  u16* A1 = (u16*)smem;
  u16* Bs = (u16*)(smem + A1N_BYTES);
  u16* A2 = (u16*)(smem + A1N_BYTES + BS_BYTES);
  float* XB = (float*)smem;

  const int tid = threadIdx.x;
  const int n0 = blockIdx.x * 32;

  // A rows: [h(0..255) | agg(256..511)] in bf16
#pragma unroll
  for (int it = 0; it < 8; ++it) {
    int linear = it * 128 + tid;            // 1024 chunks of 8 u16
    int r = linear >> 5, col = (linear & 31) * 8;
    *(uint4*)(A1 + r * A1N_STRIDE + col) =
        *(const uint4*)(h_bf + (size_t)(n0 + r) * H + col);
  }
#pragma unroll
  for (int it = 0; it < 16; ++it) {
    int linear = it * 128 + tid;            // 2048 chunks of 4 f32 -> bf16
    int r = linear >> 6, col = (linear & 63) * 4;
    float4 f = *(const float4*)(agg + (size_t)(n0 + r) * H + col);
    u16* dp = A1 + r * A1N_STRIDE + 256 + col;
    dp[0] = f2bf(f.x); dp[1] = f2bf(f.y); dp[2] = f2bf(f.z); dp[3] = f2bf(f.w);
  }
  __syncthreads();

  const int lane = tid & 31, wave = tid >> 5;
  const int m0 = lane & 15, hi = lane >> 4;
  const v8f vzero = {0.f, 0.f, 0.f, 0.f, 0.f, 0.f, 0.f, 0.f};
  v8f acc[2][4];
#pragma unroll
  for (int mt = 0; mt < 2; ++mt)
#pragma unroll
    for (int nt = 0; nt < 4; ++nt) acc[mt][nt] = vzero;

  gemm_block<512, A1N_STRIDE>(A1, Wn1_t, Bs, acc, tid);

#pragma unroll
  for (int nt = 0; nt < 4; ++nt) {
    int n = wave * 64 + nt * 16 + m0;
    float bn = b_n1[n];
#pragma unroll
    for (int mt = 0; mt < 2; ++mt)
#pragma unroll
      for (int j = 0; j < 8; ++j) {
        int m = mt * 16 + j + 8 * hi;
        A2[m * A2_STRIDE + n] = f2bf(silu_f(acc[mt][nt][j] + bn));
      }
  }
  __syncthreads();

#pragma unroll
  for (int mt = 0; mt < 2; ++mt)
#pragma unroll
    for (int nt = 0; nt < 4; ++nt) acc[mt][nt] = vzero;
  gemm_block<256, A2_STRIDE>(A2, Wn2_t, Bs, acc, tid);

#pragma unroll
  for (int nt = 0; nt < 4; ++nt) {
    int n = wave * 64 + nt * 16 + m0;
    float bn = b_n2[n];
#pragma unroll
    for (int mt = 0; mt < 2; ++mt)
#pragma unroll
      for (int j = 0; j < 8; ++j) {
        int m = mt * 16 + j + 8 * hi;
        XB[m * H + n] = acc[mt][nt][j] + bn + h[(size_t)(n0 + m) * H + n];
      }
  }
  __syncthreads();

  const int r = tid >> 2, cb = (tid & 3) * 64;
  float s = 0.f, ss = 0.f;
  const float* xr = XB + r * H;
#pragma unroll 8
  for (int c = 0; c < 64; ++c) { float x = xr[cb + c]; s += x; ss += x * x; }
  s  += __shfl_xor(s, 1, 32);  s  += __shfl_xor(s, 2, 32);
  ss += __shfl_xor(ss, 1, 32); ss += __shfl_xor(ss, 2, 32);
  float mu = s * (1.f / H);
  float rstd = __builtin_amdgcn_rsqf(ss * (1.f / H) - mu * mu + EPS);
  float* orow = h_new + (size_t)(n0 + r) * H;
#pragma unroll 4
  for (int c = 0; c < 64; ++c) {
    int cc = cb + c;
    orow[cc] = (xr[cc] - mu) * rstd * g_n[cc] + beta_n[cc];
  }
}

// ---------------- launcher ---------------------------------------------------
extern "C" void kernel_launch(void* const* d_in, const int* in_sizes, int n_in,
                              void* d_out, int out_size, void* d_ws, size_t ws_size,
                              hipStream_t stream)
{
  const float* h     = (const float*)d_in[0];
  const float* e     = (const float*)d_in[1];
  const int*   eidx  = (const int*)d_in[2];
  const float* W_src = (const float*)d_in[3];
  const float* b_src = (const float*)d_in[4];
  const float* W_dst = (const float*)d_in[5];
  const float* W_e   = (const float*)d_in[6];
  const float* W_out = (const float*)d_in[7];
  const float* b_out = (const float*)d_in[8];
  const float* W_n1  = (const float*)d_in[9];
  const float* b_n1  = (const float*)d_in[10];
  const float* W_n2  = (const float*)d_in[11];
  const float* b_n2  = (const float*)d_in[12];
  const float* g_e   = (const float*)d_in[13];
  const float* be_e  = (const float*)d_in[14];
  const float* g_n   = (const float*)d_in[15];
  const float* be_n  = (const float*)d_in[16];

  unsigned char* ws = (unsigned char*)d_ws;
  size_t off = 0;
  u16* Wcat_t = (u16*)(ws + off); off += (size_t)768 * 256 * 2;
  u16* Wout_t = (u16*)(ws + off); off += (size_t)256 * 256 * 2;
  u16* Wn1_t  = (u16*)(ws + off); off += (size_t)256 * 512 * 2;
  u16* Wn2_t  = (u16*)(ws + off); off += (size_t)256 * 256 * 2;
  u16* h_bf   = (u16*)(ws + off); off += (size_t)NNODES * H * 2;
  float* agg  = (float*)(ws + off); off += (size_t)NNODES * H * 4;

  float* h_new = (float*)d_out;
  float* e_new = (float*)d_out + (size_t)NNODES * H;

  hipMemsetAsync(agg, 0, (size_t)NNODES * H * 4, stream);
  convert_weights_kernel<<<1792, 256, 0, stream>>>(
      W_src, W_dst, W_e, W_out, W_n1, W_n2, Wcat_t, Wout_t, Wn1_t, Wn2_t);
  convert_h_kernel<<<(NNODES * H / 4) / 256, 256, 0, stream>>>(h, h_bf);
  edge_kernel<<<NEDGES / 32, 128, 0, stream>>>(
      e, eidx, h_bf, Wcat_t, Wout_t, b_src, b_out, g_e, be_e, e_new, agg);
  node_kernel<<<NNODES / 32, 128, 0, stream>>>(
      h, h_bf, agg, Wn1_t, Wn2_t, b_n1, b_n2, g_n, be_n, h_new);
}
